// _TrainedAttention_17583596109978
// MI455X (gfx1250) — matrix operand order repaired and checked by reading, not compiled, not run on hardware
//
#include <hip/hip_runtime.h>
#include <hip/hip_bf16.h>
#include <math.h>

// ---------------------------------------------------------------------------
// Types for CDNA5 WMMA bf16 path
// ---------------------------------------------------------------------------
typedef __bf16 bf16_t;
typedef bf16_t v16bf __attribute__((ext_vector_type(16)));
typedef float  v8f   __attribute__((ext_vector_type(8)));

union FragBF {
  v16bf v;
  uint4 u4[2];   // 32 bytes = two 16B chunks
};

static __device__ __forceinline__ v8f wmma_bf16(v16bf a, v16bf b, v8f c) {
  // D = A(16x32) * B(32x16) + C, f32 accumulate
  return __builtin_amdgcn_wmma_f32_16x16x32_bf16(
      /*neg_a=*/false, a, /*neg_b=*/false, b,
      /*c_mod=*/(short)0, c, /*reuse_a=*/false, /*reuse_b=*/false);
}

#define QKV 1024
#define DMODEL 1024

// ---------------------------------------------------------------------------
// f32 -> bf16 conversion (grid-stride)
// ---------------------------------------------------------------------------
__global__ __launch_bounds__(256) void cvt_f32_to_bf16(
    const float* __restrict__ in, bf16_t* __restrict__ out, int n) {
  int i = blockIdx.x * blockDim.x + threadIdx.x;
  int stride = gridDim.x * blockDim.x;
  for (; i < n; i += stride) out[i] = (bf16_t)in[i];
}

// ---------------------------------------------------------------------------
// C[M,N] = (A[M,K] * W[N,K]^T + bias[N]) * scale
// OUT_MODE 0: bf16 row-major [M][N]
// OUT_MODE 1: f32  row-major [M][N]
// OUT_MODE 2: bf16 transposed per head: out[((b*H+h)*DH+d)*S + s]
// PRESCALE: scale = 1/(8*temp[0]) (folds softmax 1/(sqrt(Dh)*T) into Q proj)
// Block = 256 threads = 8 waves (4 in M x 2 in N); wave tile 32x64;
// block tile 128x128. K stepped by 32 per WMMA.
// ---------------------------------------------------------------------------
template <int OUT_MODE, bool PRESCALE>
__global__ __launch_bounds__(256) void gemm_bf16_nt(
    const bf16_t* __restrict__ A, const bf16_t* __restrict__ W,
    const float* __restrict__ bias, const float* __restrict__ tempptr,
    void* __restrict__ outp, int M, int N, int K, int S, int H, int DH) {
  const int lane = threadIdx.x & 31;
  const int wave = threadIdx.x >> 5;
  const int hf   = lane >> 4;   // which 16-lane half
  const int l16  = lane & 15;

  const int wm = wave & 3;      // 4 waves along M
  const int wn = wave >> 2;     // 2 waves along N
  const int m0 = blockIdx.x * 128 + wm * 32;
  const int n0 = blockIdx.y * 128 + wn * 64;

  v8f c[2][4];
#pragma unroll
  for (int mi = 0; mi < 2; ++mi)
#pragma unroll
    for (int ni = 0; ni < 4; ++ni) c[mi][ni] = (v8f){};

  const bf16_t* arow0 = A + (size_t)(m0 + l16) * K;
  const bf16_t* arow1 = A + (size_t)(m0 + 16 + l16) * K;
  const bf16_t* wrow0 = W + (size_t)(n0 + l16) * K;
  const bf16_t* wrow1 = W + (size_t)(n0 + 16 + l16) * K;
  const bf16_t* wrow2 = W + (size_t)(n0 + 32 + l16) * K;
  const bf16_t* wrow3 = W + (size_t)(n0 + 48 + l16) * K;

  for (int kk = 0; kk < K; kk += 32) {
    FragBF a[2];
    a[0].u4[0] = *(const uint4*)(arow0 + kk + hf * 16);
    a[0].u4[1] = *(const uint4*)(arow0 + kk + 8 + hf * 16);
    a[1].u4[0] = *(const uint4*)(arow1 + kk + hf * 16);
    a[1].u4[1] = *(const uint4*)(arow1 + kk + 8 + hf * 16);
    FragBF b[4];
    b[0].v = *(const v16bf*)(wrow0 + kk + hf * 16);
    b[1].v = *(const v16bf*)(wrow1 + kk + hf * 16);
    b[2].v = *(const v16bf*)(wrow2 + kk + hf * 16);
    b[3].v = *(const v16bf*)(wrow3 + kk + hf * 16);
#pragma unroll
    for (int mi = 0; mi < 2; ++mi)
#pragma unroll
      for (int ni = 0; ni < 4; ++ni)
        c[mi][ni] = wmma_bf16(a[mi].v, b[ni].v, c[mi][ni]);
  }

  const float sc = PRESCALE ? (1.0f / (8.0f * tempptr[0])) : 1.0f;

#pragma unroll
  for (int mi = 0; mi < 2; ++mi) {
#pragma unroll
    for (int ni = 0; ni < 4; ++ni) {
      const int n = n0 + ni * 16 + l16;
      const float bv = bias ? bias[n] : 0.0f;
#pragma unroll
      for (int r = 0; r < 8; ++r) {
        const int m = m0 + mi * 16 + r + 8 * hf;
        const float val = (c[mi][ni][r] + bv) * sc;
        if (OUT_MODE == 0) {
          ((bf16_t*)outp)[(size_t)m * N + n] = (bf16_t)val;
        } else if (OUT_MODE == 1) {
          ((float*)outp)[(size_t)m * N + n] = val;
        } else {
          const int bIdx = m / S, s = m - bIdx * S;
          const int h = n / DH, d = n - h * DH;
          ((bf16_t*)outp)[(((size_t)bIdx * H + h) * DH + d) * S + s] =
              (bf16_t)val;
        }
      }
    }
  }
}

// ---------------------------------------------------------------------------
// Flash attention: one wave handles 16 queries of one (b,h).
// Q (pre-scaled by 1/(8*T)), K in [B*S][QKV] bf16; Vt in [(b*H+h)][DH][S];
// Ctx out [B*S][QKV] bf16. 64-key tiles: 4 score frags per iteration so the
// cross-lane softmax butterflies (ds_bpermute) are amortized over 64 keys.
// ---------------------------------------------------------------------------
__global__ __launch_bounds__(256) void flash_attn_bf16(
    const bf16_t* __restrict__ Q, const bf16_t* __restrict__ Km,
    const bf16_t* __restrict__ Vt, bf16_t* __restrict__ Ctx,
    const float* __restrict__ qscale, int S) {
  __shared__ __align__(16) bf16_t pbuf[8][16 * 64];  // per-wave P tile [q][key]

  const int lane = threadIdx.x & 31;
  const int wave = threadIdx.x >> 5;
  const int hf = lane >> 4;
  const int l16 = lane & 15;

  const int gw = blockIdx.x * 8 + wave;      // global wave id
  const int qtiles = S >> 4;                 // 128
  const int qt = gw & (qtiles - 1);
  const int bh = gw / qtiles;
  const int bIdx = bh >> 4;                  // H == 16
  const int h = bh & 15;
  const int q0 = qt * 16;

  // Q fragments (held for the whole key loop): d 0..31 and d 32..63
  const bf16_t* qrow = Q + ((size_t)(bIdx * S + q0 + l16)) * QKV + h * 64;
  FragBF qf[2];
  qf[0].u4[0] = *(const uint4*)(qrow + hf * 16);
  qf[0].u4[1] = *(const uint4*)(qrow + 8 + hf * 16);
  qf[1].u4[0] = *(const uint4*)(qrow + 32 + hf * 16);
  qf[1].u4[1] = *(const uint4*)(qrow + 40 + hf * 16);

  v8f acc[4];
#pragma unroll
  for (int ni = 0; ni < 4; ++ni) acc[ni] = (v8f){};
  float mrow[8], lsum[8];
#pragma unroll
  for (int r = 0; r < 8; ++r) { mrow[r] = -3.0e38f; lsum[r] = 0.0f; }

  bf16_t* pr = &pbuf[wave][0];

  for (int k0 = 0; k0 < S; k0 += 64) {
    // ---- scores: four 16x16 key tiles, K-dim = d (64) = 2 chained WMMAs each
    v8f s[4];
#pragma unroll
    for (int kt = 0; kt < 4; ++kt) {
      const bf16_t* krow =
          Km + ((size_t)(bIdx * S + k0 + kt * 16 + l16)) * QKV + h * 64;
      v16bf kf0 = *(const v16bf*)(krow + hf * 16);       // d 0..31
      v16bf kf1 = *(const v16bf*)(krow + 32 + hf * 16);  // d 32..63
      v8f t = wmma_bf16(qf[0].v, kf0, (v8f){});
      s[kt] = wmma_bf16(qf[1].v, kf1, t);
    }

    // ---- online softmax (rows live at (r, half); cols across 16 lanes)
    float corr[8];
#pragma unroll
    for (int r = 0; r < 8; ++r) {
      const float v0 = s[0][r], v1 = s[1][r], v2 = s[2][r], v3 = s[3][r];
      float tm = fmaxf(fmaxf(v0, v1), fmaxf(v2, v3));
      tm = fmaxf(tm, __shfl_xor(tm, 1));
      tm = fmaxf(tm, __shfl_xor(tm, 2));
      tm = fmaxf(tm, __shfl_xor(tm, 4));
      tm = fmaxf(tm, __shfl_xor(tm, 8));
      const float mnew = fmaxf(mrow[r], tm);
      const float cor = __expf(mrow[r] - mnew);
      mrow[r] = mnew;
      const float p0 = __expf(v0 - mnew);
      const float p1 = __expf(v1 - mnew);
      const float p2 = __expf(v2 - mnew);
      const float p3 = __expf(v3 - mnew);
      s[0][r] = p0; s[1][r] = p1; s[2][r] = p2; s[3][r] = p3;
      float rs = (p0 + p1) + (p2 + p3);
      rs += __shfl_xor(rs, 1);
      rs += __shfl_xor(rs, 2);
      rs += __shfl_xor(rs, 4);
      rs += __shfl_xor(rs, 8);
      lsum[r] = lsum[r] * cor + rs;
      corr[r] = cor;
    }
#pragma unroll
    for (int ni = 0; ni < 4; ++ni)
#pragma unroll
      for (int r = 0; r < 8; ++r) acc[ni][r] *= corr[r];

    // ---- redistribute P (C layout) -> A layout via wave-private LDS tile
#pragma unroll
    for (int r = 0; r < 8; ++r) {
      const int row = r + 8 * hf;
#pragma unroll
      for (int kt = 0; kt < 4; ++kt)
        pr[row * 64 + kt * 16 + l16] = (bf16_t)s[kt][r];
    }
    asm volatile("s_wait_dscnt 0" ::: "memory");  // cross-lane LDS RAW fence
    FragBF pf[2];
    pf[0].u4[0] = *(const uint4*)(pr + l16 * 64 + hf * 16);
    pf[0].u4[1] = *(const uint4*)(pr + l16 * 64 + 8 + hf * 16);
    pf[1].u4[0] = *(const uint4*)(pr + l16 * 64 + 32 + hf * 16);
    pf[1].u4[1] = *(const uint4*)(pr + l16 * 64 + 40 + hf * 16);
    asm volatile("s_wait_dscnt 0" ::: "memory");  // WAR vs next iteration

    // ---- acc += P(16x64) * V(64x64): Vt rows are [d][s], contiguous in key
#pragma unroll
    for (int t = 0; t < 2; ++t)
#pragma unroll
      for (int ni = 0; ni < 4; ++ni) {
        const bf16_t* vrow = Vt + (((size_t)bh) * 64 + ni * 16 + l16) * S +
                             k0 + t * 32 + hf * 16;
        v16bf vf = *(const v16bf*)vrow;
        acc[ni] = wmma_bf16(pf[t].v, vf, acc[ni]);
      }
  }

  // ---- normalize (q_scale folds in, post-softmax constant per row) + store
  const float qs = qscale[h];
  float invl[8];
#pragma unroll
  for (int r = 0; r < 8; ++r) invl[r] = qs / lsum[r];
#pragma unroll
  for (int ni = 0; ni < 4; ++ni)
#pragma unroll
    for (int r = 0; r < 8; ++r) {
      const int row = q0 + r + 8 * hf;
      Ctx[((size_t)(bIdx * S + row)) * QKV + h * 64 + ni * 16 + l16] =
          (bf16_t)(acc[ni][r] * invl[r]);
    }
}

// ---------------------------------------------------------------------------
// Host-side orchestration
// ---------------------------------------------------------------------------
extern "C" void kernel_launch(void* const* d_in, const int* in_sizes, int n_in,
                              void* d_out, int out_size, void* d_ws,
                              size_t ws_size, hipStream_t stream) {
  const float* x    = (const float*)d_in[0];
  const float* w_q  = (const float*)d_in[1];
  const float* b_q  = (const float*)d_in[2];
  const float* w_k  = (const float*)d_in[3];
  const float* b_k  = (const float*)d_in[4];
  const float* w_v  = (const float*)d_in[5];
  const float* b_v  = (const float*)d_in[6];
  const float* w_o  = (const float*)d_in[7];
  const float* b_o  = (const float*)d_in[8];
  const float* temp = (const float*)d_in[9];
  const float* qsc  = (const float*)d_in[10];

  const int B = 4, S = 2048, H = 16, DH = 64;
  const int M = B * S;          // 8192
  const int K = DMODEL;         // 1024
  const int N = QKV;            // 1024

  char* ws = (char*)d_ws;
  bf16_t* xb  = (bf16_t*)ws; ws += (size_t)M * K * 2;
  bf16_t* wqb = (bf16_t*)ws; ws += (size_t)N * K * 2;
  bf16_t* wkb = (bf16_t*)ws; ws += (size_t)N * K * 2;
  bf16_t* wvb = (bf16_t*)ws; ws += (size_t)N * K * 2;
  bf16_t* wob = (bf16_t*)ws; ws += (size_t)N * K * 2;
  bf16_t* Qb  = (bf16_t*)ws; ws += (size_t)M * N * 2;
  bf16_t* Kb  = (bf16_t*)ws; ws += (size_t)M * N * 2;
  bf16_t* Vtb = (bf16_t*)ws; ws += (size_t)M * N * 2;   // [B*H][64][S]
  bf16_t* Ctx = (bf16_t*)ws; ws += (size_t)M * N * 2;

  // 1) convert inputs to bf16
  cvt_f32_to_bf16<<<2048, 256, 0, stream>>>(x, xb, M * K);
  cvt_f32_to_bf16<<<512, 256, 0, stream>>>(w_q, wqb, N * K);
  cvt_f32_to_bf16<<<512, 256, 0, stream>>>(w_k, wkb, N * K);
  cvt_f32_to_bf16<<<512, 256, 0, stream>>>(w_v, wvb, N * K);
  cvt_f32_to_bf16<<<512, 256, 0, stream>>>(w_o, wob, N * K);

  // 2) QKV projections. Q is pre-scaled by 1/(8*temperature) so the flash
  //    kernel's softmax needs no per-element scaling. V stored transposed
  //    per head for the PV WMMA.
  dim3 g(M / 128, N / 128);
  gemm_bf16_nt<0, true><<<g, 256, 0, stream>>>(xb, wqb, b_q, temp, Qb, M, N, K,
                                               S, H, DH);
  gemm_bf16_nt<0, false><<<g, 256, 0, stream>>>(xb, wkb, b_k, nullptr, Kb, M,
                                                N, K, S, H, DH);
  gemm_bf16_nt<2, false><<<g, 256, 0, stream>>>(xb, wvb, b_v, nullptr, Vtb, M,
                                                N, K, S, H, DH);

  // 3) attention: B*H*(S/16) waves, 8 waves/block
  const int attn_blocks = (B * H * (S / 16)) / 8;  // 1024
  flash_attn_bf16<<<attn_blocks, 256, 0, stream>>>(Qb, Kb, Vtb, Ctx, qsc, S);

  // 4) output projection -> f32 d_out
  gemm_bf16_nt<1, false><<<g, 256, 0, stream>>>(Ctx, wob, b_o, nullptr,
                                                (float*)d_out, M, N, K, S, H,
                                                DH);
}
